// CGCNNStyle_2937757630820
// MI455X (gfx1250) — matrix operand order repaired and checked by reading, not compiled, not run on hardware
//
#include <hip/hip_runtime.h>
#include <math.h>

typedef __attribute__((ext_vector_type(16))) _Float16 v16h;
typedef __attribute__((ext_vector_type(8)))  float    v8f;

#define NN 32768
#define DD 128
#define EE 1048576
#define DEG 32
#define NG 32
#define SS 1024

// ---------------- workspace layout (bytes, 256-aligned) ----------------
#define OFF_X     ((size_t)0)                         // N*D f32  = 16 MB
#define OFF_AGG   (OFF_X    + (size_t)NN*DD*4)        // N*D f32 (agg, then x1 in-place)
#define OFF_PA    (OFF_AGG  + (size_t)NN*DD*4)        // x @ Wf_i
#define OFF_PB    (OFF_PA   + (size_t)NN*DD*4)        // x @ Wf_j
#define OFF_PC    (OFF_PB   + (size_t)NN*DD*4)        // x @ Ws_i
#define OFF_PD    (OFF_PC   + (size_t)NN*DD*4)        // x @ Ws_j
#define OFF_DIST  (OFF_PD   + (size_t)NN*DD*4)        // E f32
#define OFF_WHF   (OFF_DIST + (size_t)EE*4)           // 4*4*8*4*32*16 f16 fragments
#define OFF_PART  (OFF_WHF  + (size_t)4*4*8*4*32*16*2)// 256 blocks * 256 f32
#define OFF_STATS (OFF_PART + (size_t)256*256*4)      // mu1,rstd1,mu2,rstd2 (512 f32)
#define OFF_G     (OFF_STATS+ (size_t)512*4)          // 32*128 f32

// ---------------- prep kernels ----------------
__global__ __launch_bounds__(256) void k_embed(const int* __restrict__ z,
                                               const float* __restrict__ emb,
                                               float* __restrict__ x) {
  size_t idx = (size_t)blockIdx.x * 256 + threadIdx.x;   // N*D threads
  int n = (int)(idx >> 7), d = (int)(idx & 127);
  x[idx] = emb[(size_t)z[n] * DD + d];
}

__global__ __launch_bounds__(256) void k_dist(const int* __restrict__ ei,
                                              const float* __restrict__ pos,
                                              float* __restrict__ dist) {
  size_t e = (size_t)blockIdx.x * 256 + threadIdx.x;     // E threads
  int r = ei[e], c = ei[(size_t)EE + e];
  float dx = pos[(size_t)r*3+0] - pos[(size_t)c*3+0];
  float dy = pos[(size_t)r*3+1] - pos[(size_t)c*3+1];
  float dz = pos[(size_t)r*3+2] - pos[(size_t)c*3+2];
  dist[e] = sqrtf(dx*dx + dy*dy + dz*dz);
}

// Pre-shuffle weights into per-lane-contiguous WMMA B-fragment order (f16).
// whf[(((l*4+m)*8+nt)*4+kt)*32+lane][16] ; B frag: K = kt*32 + idx + 16*(lane/16),
// Ncol = nt*16 + lane%16.  m: 0=Wf rows0-127, 1=Wf rows128-255, 2=Ws_i, 3=Ws_j.
__global__ __launch_bounds__(256) void k_wshuffle(const float* __restrict__ Wf,
                                                  const float* __restrict__ Ws,
                                                  _Float16* __restrict__ whf) {
  size_t idx = (size_t)blockIdx.x * 256 + threadIdx.x;   // 262144 threads
  int i16 = (int)(idx & 15);  size_t r = idx >> 4;
  int lane = (int)(r & 31);   r >>= 5;
  int kt   = (int)(r & 3);    r >>= 2;
  int nt   = (int)(r & 7);    r >>= 3;
  int m    = (int)(r & 3);    r >>= 2;
  int l    = (int)r;
  int k = kt*32 + i16 + 16*(lane >> 4);
  int j = nt*16 + (lane & 15);
  const float* src = (m < 2) ? Wf : Ws;
  int krow = ((m & 1) ? 128 : 0) + k;
  whf[idx] = (_Float16)src[((size_t)l*257 + krow)*DD + j];
}

// ---------------- WMMA node GEMM: 4 products per layer ----------------
// block = 16 nodes, 8 waves; wave w owns output cols [16w,16w+16) for all 4 mats.
__global__ __launch_bounds__(256) void k_gemm(const float* __restrict__ x,
                                              const _Float16* __restrict__ whf,
                                              float* __restrict__ oA, float* __restrict__ oB,
                                              float* __restrict__ oC, float* __restrict__ oD,
                                              int layer) {
  __shared__ _Float16 tile[16 * DD];
  const int tid = threadIdx.x;
  const int n0 = blockIdx.x * 16;
  for (int i = tid; i < 16 * DD; i += 256)
    tile[i] = (_Float16)x[(size_t)n0 * DD + i];
  __syncthreads();

  const int w    = tid >> 5;
  const int lane = tid & 31;
  const int l16  = lane & 15;
  const int half = lane >> 4;

  // A fragments (contiguous K order, matches B): K = idx + 16*half
  v16h afrag[4];
  #pragma unroll
  for (int kt = 0; kt < 4; ++kt) {
    #pragma unroll
    for (int idx = 0; idx < 16; ++idx) {
      int kl = idx + 16*half;
      afrag[kt][idx] = tile[l16 * DD + kt*32 + kl];
    }
  }

  float* outs[4] = {oA, oB, oC, oD};
  #pragma unroll
  for (int m = 0; m < 4; ++m) {
    v8f c = {0.f,0.f,0.f,0.f,0.f,0.f,0.f,0.f};
    const _Float16* wb = whf + ((((size_t)(layer*4 + m)*8 + w)*4)*32)*16;
    #pragma unroll
    for (int kt = 0; kt < 4; ++kt) {
      v16h b = *(const v16h*)(wb + ((size_t)kt*32 + lane)*16);
      c = __builtin_amdgcn_wmma_f32_16x16x32_f16(false, afrag[kt], false, b,
                                                 (short)0, c, false, false);
    }
    float* o = outs[m];
    #pragma unroll
    for (int rr = 0; rr < 8; ++rr)                 // D: row = rr + 8*half, col = l16
      o[(size_t)(n0 + rr + 8*half)*DD + w*16 + l16] = c[rr];
  }
}

// ---------------- edge gate + grouped segment-sum (wave per target node) ------
__global__ __launch_bounds__(256) void k_edge(const float* __restrict__ A,
    const float* __restrict__ B, const float* __restrict__ C,
    const float* __restrict__ Dm, const int* __restrict__ erow,
    const float* __restrict__ dist, const float* __restrict__ Wf,
    const float* __restrict__ bf, const float* __restrict__ Ws,
    const float* __restrict__ bs, float* __restrict__ agg, int layer) {
  const int w = threadIdx.x >> 5, lane = threadIdx.x & 31;
  const int i = blockIdx.x * 8 + w;
  const float* wfd = Wf + ((size_t)layer*257 + 256)*DD;   // dist row of Wf
  const float* wsd = Ws + ((size_t)layer*257 + 256)*DD;
  float ai[4], ci[4], acc[4], wf4[4], ws4[4], bf4[4], bs4[4];
  #pragma unroll
  for (int q = 0; q < 4; ++q) {
    int d = q*32 + lane;
    ai[q] = A[(size_t)i*DD + d];
    ci[q] = C[(size_t)i*DD + d];
    wf4[q] = wfd[d];  ws4[q] = wsd[d];
    bf4[q] = bf[layer*DD + d];  bs4[q] = bs[layer*DD + d];
    acc[q] = 0.f;
  }
  for (int t = 0; t < DEG; ++t) {
    size_t e = (size_t)i*DEG + t;
    int j = erow[e];
    float de = dist[e];
    #pragma unroll
    for (int q = 0; q < 4; ++q) {
      int d = q*32 + lane;
      float f = ai[q] + B[(size_t)j*DD + d]  + de*wf4[q] + bf4[q];
      float s = ci[q] + Dm[(size_t)j*DD + d] + de*ws4[q] + bs4[q];
      float sig = 1.f / (1.f + __expf(-f));
      float sp  = (s > 20.f) ? s : log1pf(__expf(s));
      acc[q] += sig * sp;
    }
  }
  #pragma unroll
  for (int q = 0; q < 4; ++q)
    agg[(size_t)i*DD + q*32 + lane] = acc[q];
}

// ---------------- column stats (two-pass, deterministic, coalesced) ----------
__global__ __launch_bounds__(128) void k_stats1(const float* __restrict__ in,
                                                float* __restrict__ part) {
  const int t = threadIdx.x;
  const size_t r0 = (size_t)blockIdx.x * 128;
  float s = 0.f, ss = 0.f;
  for (int r = 0; r < 128; ++r) {
    float v = in[(r0 + r)*DD + t];
    s += v; ss += v*v;
  }
  part[(size_t)blockIdx.x*256 + t]       = s;
  part[(size_t)blockIdx.x*256 + 128 + t] = ss;
}

__global__ __launch_bounds__(128) void k_stats2(const float* __restrict__ part,
                                                float* __restrict__ stats) {
  const int t = threadIdx.x;
  float s = 0.f, ss = 0.f;
  for (int b = 0; b < 256; ++b) {
    s  += part[(size_t)b*256 + t];
    ss += part[(size_t)b*256 + 128 + t];
  }
  const float invN = 1.f / (float)NN;
  float mu = s * invN;
  float var = ss * invN - mu*mu;
  stats[t]       = mu;
  stats[128 + t] = rsqrtf(var + 1e-5f);
}

// x1 = gc*(agg-mu)*rstd + bc + x   (in-place over agg)
__global__ __launch_bounds__(256) void k_bnres(const float* __restrict__ x,
    float* __restrict__ a, const float* __restrict__ stats,
    const float* __restrict__ gc, const float* __restrict__ bc, int layer) {
  size_t idx = (size_t)blockIdx.x * 256 + threadIdx.x;
  int d = (int)(idx & 127);
  a[idx] = gc[layer*DD + d]*(a[idx] - stats[d])*stats[128 + d] + bc[layer*DD + d] + x[idx];
}

// x = relu(gn*(x1-mu)*rstd + bnb)
__global__ __launch_bounds__(256) void k_bnrelu(float* __restrict__ x,
    const float* __restrict__ x1, const float* __restrict__ stats,
    const float* __restrict__ gn, const float* __restrict__ bnb, int layer) {
  size_t idx = (size_t)blockIdx.x * 256 + threadIdx.x;
  int d = (int)(idx & 127);
  float y = gn[layer*DD + d]*(x1[idx] - stats[d])*stats[128 + d] + bnb[layer*DD + d];
  x[idx] = fmaxf(y, 0.f);
}

// ---------------- mean-pool + head ----------------
__global__ __launch_bounds__(128) void k_pool(const float* __restrict__ x,
                                              float* __restrict__ g) {
  const int t = threadIdx.x, b = blockIdx.x;
  float s = 0.f;
  for (int r = 0; r < SS; ++r) s += x[((size_t)b*SS + r)*DD + t];
  g[(size_t)b*DD + t] = s * (1.f / (float)SS);
}

__global__ __launch_bounds__(128) void k_head(const float* __restrict__ g,
    const float* __restrict__ W1, const float* __restrict__ b1,
    const float* __restrict__ W2, const float* __restrict__ b2,
    float* __restrict__ out) {
  __shared__ float h[NG * DD];
  const int t = threadIdx.x;
  for (int r = 0; r < NG; ++r) {
    float acc = b1[t];
    for (int k = 0; k < DD; ++k) acc += g[(size_t)r*DD + k] * W1[(size_t)k*DD + t];
    h[r*DD + t] = fmaxf(acc, 0.f);
  }
  __syncthreads();
  if (t < NG) {
    float acc = b2[0];
    for (int k = 0; k < DD; ++k) acc += h[t*DD + k] * W2[k];
    out[t] = acc;
  }
}

// ---------------- driver ----------------
extern "C" void kernel_launch(void* const* d_in, const int* in_sizes, int n_in,
                              void* d_out, int out_size, void* d_ws, size_t ws_size,
                              hipStream_t stream) {
  (void)in_sizes; (void)n_in; (void)out_size; (void)ws_size;
  const int*   z    = (const int*)  d_in[0];
  const float* pos  = (const float*)d_in[1];
  // d_in[2] = batch: graphs are contiguous blocks of 1024 nodes; not needed.
  const int*   ei   = (const int*)  d_in[3];
  const float* emb  = (const float*)d_in[4];
  const float* Wf   = (const float*)d_in[5];
  const float* bf   = (const float*)d_in[6];
  const float* Ws   = (const float*)d_in[7];
  const float* bs   = (const float*)d_in[8];
  const float* gc   = (const float*)d_in[9];
  const float* bc   = (const float*)d_in[10];
  const float* gn   = (const float*)d_in[11];
  const float* bnb  = (const float*)d_in[12];
  const float* W1   = (const float*)d_in[13];
  const float* b1   = (const float*)d_in[14];
  const float* W2   = (const float*)d_in[15];
  const float* b2   = (const float*)d_in[16];

  char* ws = (char*)d_ws;
  float*    x     = (float*)   (ws + OFF_X);
  float*    agg   = (float*)   (ws + OFF_AGG);
  float*    pA    = (float*)   (ws + OFF_PA);
  float*    pB    = (float*)   (ws + OFF_PB);
  float*    pC    = (float*)   (ws + OFF_PC);
  float*    pD    = (float*)   (ws + OFF_PD);
  float*    dist  = (float*)   (ws + OFF_DIST);
  _Float16* whf   = (_Float16*)(ws + OFF_WHF);
  float*    part  = (float*)   (ws + OFF_PART);
  float*    stats = (float*)   (ws + OFF_STATS);
  float*    g     = (float*)   (ws + OFF_G);

  k_embed   <<<(NN*DD)/256, 256, 0, stream>>>(z, emb, x);
  k_dist    <<<EE/256,      256, 0, stream>>>(ei, pos, dist);
  k_wshuffle<<<(4*4*8*4*32*16)/256, 256, 0, stream>>>(Wf, Ws, whf);

  for (int l = 0; l < 4; ++l) {
    k_gemm  <<<NN/16,        256, 0, stream>>>(x, whf, pA, pB, pC, pD, l);
    k_edge  <<<NN/8,         256, 0, stream>>>(pA, pB, pC, pD, ei, dist,
                                               Wf, bf, Ws, bs, agg, l);
    k_stats1<<<256,          128, 0, stream>>>(agg, part);
    k_stats2<<<1,            128, 0, stream>>>(part, stats);
    k_bnres <<<(NN*DD)/256,  256, 0, stream>>>(x, agg, stats, gc, bc, l);
    k_stats1<<<256,          128, 0, stream>>>(agg, part);
    k_stats2<<<1,            128, 0, stream>>>(part, stats + 256);
    k_bnrelu<<<(NN*DD)/256,  256, 0, stream>>>(x, agg, stats + 256, gn, bnb, l);
  }

  k_pool<<<NG, 128, 0, stream>>>(x, g);
  k_head<<<1,  128, 0, stream>>>(g, W1, b1, W2, b2, (float*)d_out);
}
